// LatentEncoder2_76192719831700
// MI455X (gfx1250) — matrix operand order repaired and checked by reading, not compiled, not run on hardware
//
#include <hip/hip_runtime.h>
#include <hip/hip_bf16.h>
#include <math.h>

#define KK 30

typedef __attribute__((ext_vector_type(16))) _Float16 v16h;
typedef __attribute__((ext_vector_type(8)))  float    v8f;
typedef __attribute__((ext_vector_type(2)))  _Float16 h2v;

// ======================= WMMA GEMM: C = act(A @ Bt^T + bias) =======================
// One wave computes a 16-row x Npad strip: A fragment loaded once per K-step,
// NT back-to-back v_wmma_f32_16x16x32_f16 with independent accumulators.
// A: MxKd row-major f16 (Kd mult of 32, M mult of 16). Bt: Npad x Kd f16 (B^T, zero-padded).
// flags bit0: SiLU, bit1: f16 output (else f32). Columns >= Nreal written as 0.
template <int NT>
__global__ void k_gemm_t(const _Float16* __restrict__ A, const _Float16* __restrict__ Bt,
                         const float* __restrict__ bias, void* __restrict__ Cout,
                         int M, int Nreal, int Kd, int flags)
{
    constexpr int Npad = NT * 16;
    int wave = blockIdx.x * (blockDim.x >> 5) + (threadIdx.x >> 5);
    if (wave >= (M >> 4)) return;               // wave-uniform: EXEC stays all-ones
    int lane = threadIdx.x & 31;
    int row = lane & 15, hsel = lane >> 4;
    const _Float16* Ab = A  + (size_t)(wave * 16 + row) * Kd;
    const _Float16* Bb = Bt + (size_t)row * Kd;
    v8f acc[NT];
#pragma unroll
    for (int t = 0; t < NT; ++t) acc[t] = (v8f){};
    for (int k0 = 0; k0 < Kd; k0 += 32) {
        v16h af;
#pragma unroll
        for (int v = 0; v < 8; ++v) {
            // A 16x32 f16 layout: VGPR v holds K pairs interleaved by lane-half (ISA 7.12.2)
            int ka = (v < 4 ? 2 * v : 16 + 2 * (v - 4)) + 8 * hsel;
            h2v pa = *(const h2v*)(Ab + k0 + ka);
            af[2 * v] = pa.x; af[2 * v + 1] = pa.y;
        }
#pragma unroll
        for (int t = 0; t < NT; ++t) {
            v16h bf;
#pragma unroll
            for (int v = 0; v < 8; ++v) {
                // B 32x16 f16 layout: same K order as A (pairs interleaved by lane-half)
                int kb = (v < 4 ? 2 * v : 16 + 2 * (v - 4)) + 8 * hsel;
                h2v pb = *(const h2v*)(Bb + ((size_t)t * 16) * Kd + k0 + kb);
                bf[2 * v] = pb.x; bf[2 * v + 1] = pb.y;
            }
            acc[t] = __builtin_amdgcn_wmma_f32_16x16x32_f16(false, af, false, bf,
                                                            (short)0, acc[t], false, false);
        }
    }
#pragma unroll
    for (int t = 0; t < NT; ++t) {
        int ncol = t * 16 + row;
        bool keep = ncol < Nreal;
        float bval = (bias != nullptr && keep) ? bias[ncol] : 0.0f;
#pragma unroll
        for (int r = 0; r < 8; ++r) {
            float v = acc[t][r] + bval;
            if (flags & 1) v = v / (1.0f + __expf(-v));   // SiLU (wave-uniform branch)
            v = keep ? v : 0.0f;
            size_t idx = (size_t)(wave * 16 + r + 8 * hsel) * Npad + ncol;
            if (flags & 2) ((_Float16*)Cout)[idx] = (_Float16)v;
            else           ((float*)   Cout)[idx] = v;
        }
    }
}

// ======================= weight prep: Bt[n][k] = W[k][n], f16, zero-padded =======================
__global__ void k_prep_bt(const float* __restrict__ W, _Float16* __restrict__ Bt,
                          int Kin, int Nin, int Kpad, int Npad)
{
    int i = blockIdx.x * blockDim.x + threadIdx.x;
    if (i >= Kpad * Npad) return;
    int n = i / Kpad, k = i - n * Kpad;
    float v = (k < Kin && n < Nin) ? W[(size_t)k * Nin + n] : 0.0f;
    Bt[(size_t)n * Kpad + k] = (_Float16)v;
}

// ======================= kNN graph (top-K by squared distance, same batch) =======================
__global__ void k_knn(const float* __restrict__ X, const int* __restrict__ batch,
                      int* __restrict__ srcIdx, int N)
{
    int n = blockIdx.x * blockDim.x + threadIdx.x;
    if (n >= N) return;
    float bd[KK]; int bi[KK];
    for (int t = 0; t < KK; ++t) { bd[t] = 3e38f; bi[t] = 0; }
    float px = X[n*3], py = X[n*3+1], pz = X[n*3+2];
    int bn = batch[n];
    for (int j = 0; j < N; ++j) {
        if (j == n || batch[j] != bn) continue;
        float dx = px - X[j*3], dy = py - X[j*3+1], dz = pz - X[j*3+2];
        float d2 = dx*dx + dy*dy + dz*dz;
        if (d2 < bd[KK-1]) {
            int p = KK - 1;
            while (p > 0 && bd[p-1] > d2) { bd[p] = bd[p-1]; bi[p] = bi[p-1]; --p; }
            bd[p] = d2; bi[p] = j;
        }
    }
    for (int t = 0; t < KK; ++t) srcIdx[n*KK + t] = bi[t];
}

// ======================= edge features: RBF + pos-emb + rotation frames =======================
__global__ void k_edge_geom(const float* __restrict__ X, const int* __restrict__ src,
                            float* __restrict__ ef, float* __restrict__ R, int N)
{
    int E = N * KK;
    int e = blockIdx.x * blockDim.x + threadIdx.x;
    if (e >= E) return;
    int n = e / KK, s = src[e];
    float vx = X[n*3]-X[s*3], vy = X[n*3+1]-X[s*3+1], vz = X[n*3+2]-X[s*3+2];
    float d = sqrtf(vx*vx + vy*vy + vz*vz);
    float* efe = ef + (size_t)e * 32;
    for (int t = 0; t < 16; ++t) {
        float mu = (20.0f/15.0f) * (float)t;
        float z = (d - mu) * 0.8f;          // sigma = 20/16 = 1.25
        efe[t] = __expf(-z*z);
    }
    float dd = (float)(s - n);
    for (int t = 0; t < 8; ++t) {
        float fr = __expf(-(float)(2*t) * (9.210340371976184f / 16.0f));
        float ang = dd * fr;
        efe[16+t] = cosf(ang);
        efe[24+t] = sinf(ang);
    }
    float inl = 1.0f / (d + 1e-8f);
    float zx = vx*inl, zy = vy*inl, zz = vz*inl;
    float rx, ry, rz;
    if (fabsf(zx) < 0.99f) { rx = 1.f; ry = 0.f; rz = 0.f; } else { rx = 0.f; ry = 1.f; rz = 0.f; }
    float xx = ry*zz - rz*zy, xy = rz*zx - rx*zz, xz = rx*zy - ry*zx;
    float xn = 1.0f / (sqrtf(xx*xx + xy*xy + xz*xz) + 1e-8f);
    xx *= xn; xy *= xn; xz *= xn;
    float yx = zy*xz - zz*xy, yy = zz*xx - zx*xz, yz = zx*xy - zy*xx;
    float* Re = R + (size_t)e * 9;
    Re[0]=xx; Re[1]=xy; Re[2]=xz;
    Re[3]=yx; Re[4]=yy; Re[5]=yz;
    Re[6]=zx; Re[7]=zy; Re[8]=zz;
}

// ======================= node embeddings =======================
__device__ inline void vunit3(float* v){
    float i = 1.0f / (sqrtf(v[0]*v[0]+v[1]*v[1]+v[2]*v[2]) + 1e-8f);
    v[0]*=i; v[1]*=i; v[2]*=i;
}
__device__ inline void vcross3(const float* a, const float* b, float* o){
    o[0]=a[1]*b[2]-a[2]*b[1]; o[1]=a[2]*b[0]-a[0]*b[2]; o[2]=a[0]*b[1]-a[1]*b[0];
}
__device__ inline float vdot3(const float* a, const float* b){ return a[0]*b[0]+a[1]*b[1]+a[2]*b[2]; }
__device__ inline void fetch_atom(const float* bb, int t, float* o){
    int n = t / 3, j = t - n * 3;
    const float* p = bb + ((size_t)n*4 + j)*3;
    o[0]=p[0]; o[1]=p[1]; o[2]=p[2];
}
__device__ float dihedral_at(const float* bb, int idx, int N){
    if (idx == 0 || idx >= 3*N - 2) return 0.0f;   // pad (1,2)
    int r = idx - 1;
    float a[3],b[3],c[3],d[3];
    fetch_atom(bb,r,a); fetch_atom(bb,r+1,b); fetch_atom(bb,r+2,c); fetch_atom(bb,r+3,d);
    float u2[3]={b[0]-a[0],b[1]-a[1],b[2]-a[2]};
    float u1[3]={c[0]-b[0],c[1]-b[1],c[2]-b[2]};
    float u0[3]={d[0]-c[0],d[1]-c[1],d[2]-c[2]};
    vunit3(u2); vunit3(u1); vunit3(u0);
    float n2[3], n1[3];
    vcross3(u2,u1,n2); vunit3(n2);
    vcross3(u1,u0,n1); vunit3(n1);
    float cd = vdot3(n2,n1);
    cd = fminf(fmaxf(cd, -1.0f + 1e-7f), 1.0f - 1e-7f);
    float s = vdot3(u2,n1);
    float sg = (s > 0.0f) ? 1.0f : ((s < 0.0f) ? -1.0f : 0.0f);
    return sg * acosf(cd);
}
__global__ void k_bb_emb(const float* __restrict__ X, const float* __restrict__ bb,
                         float* __restrict__ bbemb, int N)
{
    int n = blockIdx.x * blockDim.x + threadIdx.x;
    if (n >= N) return;
    float D0 = dihedral_at(bb, 3*n,   N);
    float D1 = dihedral_at(bb, 3*n+1, N);
    float D2 = dihedral_at(bb, 3*n+2, N);
    float* o = bbemb + (size_t)n * 28;          // (4,7)
    o[0]=cosf(D0); o[1]=cosf(D1); o[2]=cosf(D2);
    o[3]=sinf(D0); o[4]=sinf(D1); o[5]=sinf(D2);
    o[6]=0.0f;
    const float* xc = X + (size_t)n * 3;
    float l1[7][3];
    for (int j = 0; j < 4; ++j)
        for (int d = 0; d < 3; ++d)
            l1[j][d] = bb[((size_t)n*4 + j)*3 + d] - xc[d];
    float fwd[3]={0,0,0}, bwd[3]={0,0,0};
    if (n < N-1){ for(int d=0;d<3;++d) fwd[d]=X[(size_t)(n+1)*3+d]-xc[d]; vunit3(fwd); }
    if (n > 0)  { for(int d=0;d<3;++d) bwd[d]=X[(size_t)(n-1)*3+d]-xc[d]; vunit3(bwd); }
    for (int d=0; d<3; ++d){ l1[4][d]=fwd[d]; l1[5][d]=bwd[d]; }
    float nn[3], ca[3], cc[3];
    for (int d=0; d<3; ++d){
        nn[d]=bb[((size_t)n*4+0)*3+d]; ca[d]=bb[((size_t)n*4+1)*3+d]; cc[d]=bb[((size_t)n*4+2)*3+d];
    }
    float bv[3]={ca[0]-nn[0],ca[1]-nn[1],ca[2]-nn[2]};
    float cv[3]={cc[0]-ca[0],cc[1]-ca[1],cc[2]-ca[2]};
    float av[3]; vcross3(bv,cv,av);
    for (int d=0; d<3; ++d)
        l1[6][d] = -0.58273431f*av[d] + 0.56802827f*bv[d] - 0.54067466f*cv[d] + ca[d] - xc[d];
    for (int d=0; d<3; ++d)
        for (int j=0; j<7; ++j){
            float v = l1[j][d];
            if (!(v == v)) v = 0.0f;            // nan_to_num
            o[(1+d)*7 + j] = v;
        }
}
__global__ void k_atom_emb(const float* __restrict__ a91, float* __restrict__ atemb, int N)
{
    int i = blockIdx.x * blockDim.x + threadIdx.x;
    if (i >= N*91) return;
    int n = i / 91, j = i - n * 91;
    atemb[((size_t)n*4 + 0)*91 + j] = 0.0f;
    for (int d = 0; d < 3; ++d)
        atemb[((size_t)n*4 + 1 + d)*91 + j] = a91[((size_t)n*91 + j)*3 + d];
}

__global__ void k_cvt16(const float* __restrict__ s, _Float16* __restrict__ d, int cnt)
{
    int i = blockIdx.x * blockDim.x + threadIdx.x;
    if (i < cnt) d[i] = (_Float16)s[i];
}

// ======================= build rotated (optionally gated) edge features, f16, padded =======================
__global__ void k_build_xrg(const float* __restrict__ emb, int cin, int cpad,
                            const float* __restrict__ g, const int* __restrict__ src,
                            const float* __restrict__ R,
                            _Float16* __restrict__ o0, _Float16* __restrict__ o13, int E)
{
    int i = blockIdx.x * blockDim.x + threadIdx.x;
    if (i >= E * cpad) return;
    int e = i / cpad, j = i - e * cpad;
    if (j >= cin) {
        o0[(size_t)e*cpad + j] = (_Float16)0.0f;
        o13[((size_t)e*3+0)*cpad + j] = (_Float16)0.0f;
        o13[((size_t)e*3+1)*cpad + j] = (_Float16)0.0f;
        o13[((size_t)e*3+2)*cpad + j] = (_Float16)0.0f;
        return;
    }
    int s = src[e];
    float gg = g ? g[(size_t)e*cpad + j] : 1.0f;
    float r0 = emb[((size_t)s*4 + 0)*cin + j];
    float m0 = emb[((size_t)s*4 + 1)*cin + j];
    float m1 = emb[((size_t)s*4 + 2)*cin + j];
    float m2 = emb[((size_t)s*4 + 3)*cin + j];
    const float* Re = R + (size_t)e * 9;
    o0[(size_t)e*cpad + j] = (_Float16)(r0 * gg);
    o13[((size_t)e*3+0)*cpad + j] = (_Float16)((Re[0]*m0 + Re[1]*m1 + Re[2]*m2) * gg);
    o13[((size_t)e*3+1)*cpad + j] = (_Float16)((Re[3]*m0 + Re[4]*m1 + Re[5]*m2) * gg);
    o13[((size_t)e*3+2)*cpad + j] = (_Float16)((Re[6]*m0 + Re[7]*m1 + Re[8]*m2) * gg);
}

// ======================= SO(3)-conv reduction: x[n] = mean_k D^T y =======================
__global__ void k_so3_reduce(const _Float16* __restrict__ y0, const _Float16* __restrict__ y13,
                             const float* __restrict__ R, float* __restrict__ x, int off, int N)
{
    int i = blockIdx.x * blockDim.x + threadIdx.x;
    if (i >= N*32) return;
    int n = i / 32, c = i - n * 32;
    float a0=0, a1=0, a2=0, a3=0;
    for (int k = 0; k < KK; ++k) {
        size_t e = (size_t)n*KK + k;
        const float* Re = R + e * 9;
        a0 += (float)y0[e*32 + c];
        float y1 = (float)y13[(e*3+0)*32 + c];
        float y2 = (float)y13[(e*3+1)*32 + c];
        float y3 = (float)y13[(e*3+2)*32 + c];
        a1 += Re[0]*y1 + Re[3]*y2 + Re[6]*y3;   // R^T
        a2 += Re[1]*y1 + Re[4]*y2 + Re[7]*y3;
        a3 += Re[2]*y1 + Re[5]*y2 + Re[8]*y3;
    }
    float s = 1.0f / (float)KK;
    x[((size_t)n*4+0)*64 + off + c] = a0*s;
    x[((size_t)n*4+1)*64 + off + c] = a1*s;
    x[((size_t)n*4+2)*64 + off + c] = a2*s;
    x[((size_t)n*4+3)*64 + off + c] = a3*s;
}

// ======================= invariant message input: [x[src,0], x[dst,0], ef] f16 =======================
__global__ void k_build_inv(const float* __restrict__ x, const float* __restrict__ ef,
                            const int* __restrict__ src, _Float16* __restrict__ inv, int N)
{
    int E = N * KK;
    int i = blockIdx.x * blockDim.x + threadIdx.x;
    if (i >= E * 160) return;
    int e = i / 160, j = i - e * 160;
    float v;
    if      (j <  64) v = x[((size_t)src[e]*4)*64 + j];
    else if (j < 128) v = x[((size_t)(e/KK)*4)*64 + (j-64)];
    else              v = ef[(size_t)e*32 + (j-128)];
    inv[i] = (_Float16)v;
}

__global__ void k_softmax(const float* __restrict__ logits, float* __restrict__ alpha, int N)
{
    int i = blockIdx.x * blockDim.x + threadIdx.x;
    if (i >= N*8) return;
    int n = i / 8, h = i - n * 8;
    float mx = -3e38f;
    for (int k = 0; k < KK; ++k)
        mx = fmaxf(mx, logits[((size_t)n*KK + k)*16 + h]);
    float sum = 0.0f;
    for (int k = 0; k < KK; ++k) {
        float v = __expf(logits[((size_t)n*KK + k)*16 + h] - mx);
        alpha[((size_t)n*KK + k)*8 + h] = v;
        sum += v;
    }
    float inv = 1.0f / sum;
    for (int k = 0; k < KK; ++k)
        alpha[((size_t)n*KK + k)*8 + h] *= inv;
}

// agg[n,l,c] = sum_k alpha[n,k,head(c)] * (D^T v)[l,c]
__global__ void k_aggregate(const _Float16* __restrict__ v0, const _Float16* __restrict__ v13,
                            const float* __restrict__ alpha, const float* __restrict__ R,
                            _Float16* __restrict__ agg0, _Float16* __restrict__ agg13, int N)
{
    int i = blockIdx.x * blockDim.x + threadIdx.x;
    if (i >= N*64) return;
    int n = i / 64, c = i - n * 64;
    int h = c >> 3;
    float a0=0, a1=0, a2=0, a3=0;
    for (int k = 0; k < KK; ++k) {
        size_t e = (size_t)n*KK + k;
        float al = alpha[e*8 + h];
        const float* Re = R + e * 9;
        a0 += al * (float)v0[e*64 + c];
        float y1 = (float)v13[(e*3+0)*64 + c];
        float y2 = (float)v13[(e*3+1)*64 + c];
        float y3 = (float)v13[(e*3+2)*64 + c];
        a1 += al*(Re[0]*y1 + Re[3]*y2 + Re[6]*y3);
        a2 += al*(Re[1]*y1 + Re[4]*y2 + Re[7]*y3);
        a3 += al*(Re[2]*y1 + Re[5]*y2 + Re[8]*y3);
    }
    agg0[(size_t)n*64 + c] = (_Float16)a0;
    agg13[((size_t)n*3+0)*64 + c] = (_Float16)a1;
    agg13[((size_t)n*3+1)*64 + c] = (_Float16)a2;
    agg13[((size_t)n*3+2)*64 + c] = (_Float16)a3;
}

__global__ void k_add_out(float* __restrict__ x, const float* __restrict__ o0,
                          const float* __restrict__ o13, int N)
{
    int i = blockIdx.x * blockDim.x + threadIdx.x;
    if (i >= N*256) return;
    int n = i >> 8, rem = i & 255;
    int l = rem >> 6, c = rem & 63;
    float add = (l == 0) ? o0[(size_t)n*64 + c] : o13[((size_t)n*3 + (l-1))*64 + c];
    x[i] += add;
}

__global__ void k_x0f16(const float* __restrict__ x, _Float16* __restrict__ o, int N)
{
    int i = blockIdx.x * blockDim.x + threadIdx.x;
    if (i >= N*64) return;
    o[i] = (_Float16)x[(size_t)(i >> 6)*256 + (i & 63)];
}

__global__ void k_gate(const float* __restrict__ x, const float* __restrict__ g,
                       _Float16* __restrict__ h0, _Float16* __restrict__ h13, int N)
{
    int i = blockIdx.x * blockDim.x + threadIdx.x;
    if (i >= N*256) return;
    int n = i >> 8, rem = i & 255;
    int l = rem >> 6, c = rem & 63;
    float v = x[i] * g[(size_t)n*64 + c];
    if (l == 0) h0[(size_t)n*64 + c] = (_Float16)v;
    else        h13[((size_t)n*3 + (l-1))*64 + c] = (_Float16)v;
}

__global__ void k_final(float* __restrict__ dout, const float* __restrict__ o0,
                        const float* __restrict__ o13, int which, int N)
{
    int i = blockIdx.x * blockDim.x + threadIdx.x;
    if (i >= N*128) return;
    int n = i >> 7, rem = i & 127;
    int l = rem >> 5, c = rem & 31;
    float v = (l == 0) ? o0[(size_t)n*32 + c] : o13[((size_t)n*3 + (l-1))*32 + c];
    dout[((size_t)which*N + n)*128 + rem] = v;
}

// =============================================================================================
extern "C" void kernel_launch(void* const* d_in, const int* in_sizes, int n_in,
                              void* d_out, int out_size, void* d_ws, size_t ws_size,
                              hipStream_t stream)
{
    const float* X   = (const float*)d_in[0];
    const float* bb  = (const float*)d_in[1];
    const float* a91 = (const float*)d_in[2];
    const int* batch = (const int*)d_in[4];
    int N   = in_sizes[0] / 3;
    int E   = N * KK;
    int E3v = E * 3;
    int N3v = N * 3;
    (void)n_in; (void)out_size; (void)ws_size;

    // -------- workspace carve-up --------
    char* base = (char*)d_ws; size_t off = 0;
    auto take = [&](size_t bytes)->char* {
        char* p = base + off; off += (bytes + 255) & ~(size_t)255; return p;
    };
    int*      srcIdx = (int*)     take((size_t)E * 4);
    float*    R      = (float*)   take((size_t)E * 9 * 4);
    float*    ef32   = (float*)   take((size_t)E * 32 * 4);
    _Float16* ef16   = (_Float16*)take((size_t)E * 32 * 2);
    float*    bbemb  = (float*)   take((size_t)N * 28 * 4);
    float*    atemb  = (float*)   take((size_t)N * 364 * 4);
    float*    x      = (float*)   take((size_t)N * 256 * 4);
    float*    alpha  = (float*)   take((size_t)E * 8 * 4);
    _Float16* agg0   = (_Float16*)take((size_t)N * 64 * 2);
    _Float16* agg13  = (_Float16*)take((size_t)N3v * 64 * 2);
    _Float16* x0f    = (_Float16*)take((size_t)N * 64 * 2);
    _Float16* h0     = (_Float16*)take((size_t)N * 64 * 2);
    _Float16* h13    = (_Float16*)take((size_t)N3v * 64 * 2);
    float*    out0   = (float*)   take((size_t)N * 64 * 4);
    float*    out13  = (float*)   take((size_t)N3v * 64 * 4);
    float*    gbuf   = (float*)   take((size_t)N * 64 * 4);
    char*     bufA   = take((size_t)E3v * 96 * 2);   // inv / xr13 / xrg13
    char*     bufB   = take((size_t)E * 96 * 2);     // m / xr0 / t / xrg0
    char*     bufC   = take((size_t)E * 96 * 4);     // g / logits / v0 / y13
    char*     bufD   = take((size_t)E3v * 64 * 2);   // v13 / y0
    char*     warena = take((size_t)2 * 1024 * 1024);

    // -------- weight prep (f16 transposed, padded) --------
    size_t woff = 0;
    auto prep = [&](const float* W, int Kin, int Nin, int Kpad, int Npad)->_Float16* {
        _Float16* bt = (_Float16*)(warena + woff);
        woff += ((size_t)Kpad * Npad * 2 + 255) & ~(size_t)255;
        int tot = Kpad * Npad;
        k_prep_bt<<<(tot + 255) / 256, 256, 0, stream>>>(W, bt, Kin, Nin, Kpad, Npad);
        return bt;
    };
    auto F = [&](int idx)->const float* { return (const float*)d_in[idx]; };

    // params leaves (jax sorted-key flatten order)
    _Float16* bt_aw1 = prep(F(7),  32, 64, 32, 64);
    _Float16* bt_aw2 = prep(F(8),  64, 91, 64, 96);
    _Float16* bt_al0 = prep(F(9),  91, 32, 96, 32);
    _Float16* bt_al1 = prep(F(10), 91, 32, 96, 32);
    _Float16* bt_bw1 = prep(F(13), 32, 64, 32, 64);
    _Float16* bt_bw2 = prep(F(14), 64, 7,  64, 32);
    _Float16* bt_bl0 = prep(F(15), 7,  32, 32, 32);
    _Float16* bt_bl1 = prep(F(16), 7,  32, 32, 32);

    _Float16 *bt_msg[6], *bt_alp[6], *bt_v0[6], *bt_v1[6], *bt_o0[6], *bt_o1[6];
    _Float16 *bt_g[4], *bt_f0[4], *bt_f1[4], *bt_ew1[4], *bt_ew2[4];
    const float *b_msg[6], *b_g[4], *eb1[4], *eb2[4];
    for (int L = 0; L < 4; ++L) {
        int p = 33 + L * 11;     // b_g,b_msg,w_alpha,w_f0,w_f1,w_g,w_msg,w_o0,w_o1,w_v0,w_v1
        b_g[L]    = F(p + 0);
        b_msg[L]  = F(p + 1);
        bt_alp[L] = prep(F(p + 2), 32, 8, 32, 16);
        bt_f0[L]  = prep(F(p + 3), 64, 64, 64, 64);
        bt_f1[L]  = prep(F(p + 4), 64, 64, 64, 64);
        bt_g[L]   = prep(F(p + 5), 64, 64, 64, 64);
        bt_msg[L] = prep(F(p + 6), 160, 32, 160, 32);
        bt_o0[L]  = prep(F(p + 7), 64, 64, 64, 64);
        bt_o1[L]  = prep(F(p + 8), 64, 64, 64, 64);
        bt_v0[L]  = prep(F(p + 9), 64, 64, 64, 64);
        bt_v1[L]  = prep(F(p + 10), 64, 64, 64, 64);
        int q = 17 + L * 4;      // b1,b2,w1,w2
        eb1[L]    = F(q + 0);
        eb2[L]    = F(q + 1);
        bt_ew1[L] = prep(F(q + 2), 160, 64, 160, 64);
        bt_ew2[L] = prep(F(q + 3), 64, 32, 64, 32);
    }
    int bases[2] = {84, 77};     // mu, logvar
    for (int t = 0; t < 2; ++t) {
        int li = 4 + t, p = bases[t];   // b_msg,w_alpha,w_msg,w_o0,w_o1,w_v0,w_v1
        b_msg[li]  = F(p + 0);
        bt_alp[li] = prep(F(p + 1), 32, 8, 32, 16);
        bt_msg[li] = prep(F(p + 2), 160, 32, 160, 32);
        bt_o0[li]  = prep(F(p + 3), 64, 32, 64, 32);
        bt_o1[li]  = prep(F(p + 4), 64, 32, 64, 32);
        bt_v0[li]  = prep(F(p + 5), 64, 64, 64, 64);
        bt_v1[li]  = prep(F(p + 6), 64, 64, 64, 64);
    }

    // strip-mined WMMA GEMM dispatch (one wave = 16 rows x Npad)
    auto gemm = [&](const void* A, const _Float16* Bt, const float* bias, void* C,
                    int M, int Npad, int Nreal, int Kd, int flags) {
        int mtiles = M / 16;
        int blocks = (mtiles + 3) / 4;
        const _Float16* Ah = (const _Float16*)A;
        switch (Npad / 16) {
            case 1: k_gemm_t<1><<<blocks, 128, 0, stream>>>(Ah, Bt, bias, C, M, Nreal, Kd, flags); break;
            case 2: k_gemm_t<2><<<blocks, 128, 0, stream>>>(Ah, Bt, bias, C, M, Nreal, Kd, flags); break;
            case 4: k_gemm_t<4><<<blocks, 128, 0, stream>>>(Ah, Bt, bias, C, M, Nreal, Kd, flags); break;
            default: k_gemm_t<6><<<blocks, 128, 0, stream>>>(Ah, Bt, bias, C, M, Nreal, Kd, flags); break;
        }
    };

    // -------- stage 0: graph + geometry + node embeddings --------
    k_knn<<<(N + 63) / 64, 64, 0, stream>>>(X, batch, srcIdx, N);
    k_edge_geom<<<(E + 255) / 256, 256, 0, stream>>>(X, srcIdx, ef32, R, N);
    k_cvt16<<<((E * 32) + 255) / 256, 256, 0, stream>>>(ef32, ef16, E * 32);
    k_bb_emb<<<(N + 127) / 128, 128, 0, stream>>>(X, bb, bbemb, N);
    k_atom_emb<<<((N * 91) + 255) / 256, 256, 0, stream>>>(a91, atemb, N);

    // -------- SO(3) convolutions -> initial x (N,4,64) --------
    auto so3 = [&](const float* emb, int cin, int cpad, const float* b1, const float* b2,
                   _Float16* btW1, _Float16* btW2, _Float16* btL0, _Float16* btL1, int coff) {
        gemm(ef16, btW1, b1, bufB, E, 64, 64, 32, 1 | 2);                  // t = silu(ef@w1+b1), f16
        gemm(bufB, btW2, b2, bufC, E, cpad, cin, 64, 0);                   // g  = t@w2+b2, f32
        int tot = E * cpad;
        k_build_xrg<<<(tot + 255) / 256, 256, 0, stream>>>(emb, cin, cpad, (const float*)bufC,
                                                           srcIdx, R, (_Float16*)bufB,
                                                           (_Float16*)bufA, E);
        gemm(bufB, btL0, nullptr, bufD, E, 32, 32, cpad, 2);               // y0  f16
        gemm(bufA, btL1, nullptr, bufC, E3v, 32, 32, cpad, 2);             // y13 f16
        k_so3_reduce<<<((N * 32) + 255) / 256, 256, 0, stream>>>(
            (const _Float16*)bufD, (const _Float16*)bufC, R, x, coff, N);
    };
    so3(bbemb, 7,  32, F(11), F(12), bt_bw1, bt_bw2, bt_bl0, bt_bl1, 0);
    so3(atemb, 91, 96, F(5),  F(6),  bt_aw1, bt_aw2, bt_al0, bt_al1, 32);

    // -------- attention / FFN / edge-update layers --------
    auto attn = [&](int li, int outc) {
        int tot = E * 160;
        k_build_inv<<<(tot + 255) / 256, 256, 0, stream>>>(x, ef32, srcIdx, (_Float16*)bufA, N);
        gemm(bufA, bt_msg[li], b_msg[li], bufB, E, 32, 32, 160, 1 | 2);    // m f16 (silu)
        gemm(bufB, bt_alp[li], nullptr, bufC, E, 16, 8, 32, 0);            // logits f32
        k_softmax<<<((N * 8) + 255) / 256, 256, 0, stream>>>((const float*)bufC, alpha, N);
        tot = E * 64;
        k_build_xrg<<<(tot + 255) / 256, 256, 0, stream>>>(x, 64, 64, nullptr, srcIdx, R,
                                                           (_Float16*)bufB, (_Float16*)bufA, E);
        gemm(bufB, bt_v0[li], nullptr, bufC, E, 64, 64, 64, 2);            // v0  f16
        gemm(bufA, bt_v1[li], nullptr, bufD, E3v, 64, 64, 64, 2);          // v13 f16
        k_aggregate<<<((N * 64) + 255) / 256, 256, 0, stream>>>(
            (const _Float16*)bufC, (const _Float16*)bufD, alpha, R, agg0, agg13, N);
        gemm(agg0,  bt_o0[li], nullptr, out0,  N,   outc, outc, 64, 0);
        gemm(agg13, bt_o1[li], nullptr, out13, N3v, outc, outc, 64, 0);
    };
    auto ffn = [&](int li) {
        k_add_out<<<((N * 256) + 255) / 256, 256, 0, stream>>>(x, out0, out13, N);
        k_x0f16<<<((N * 64) + 255) / 256, 256, 0, stream>>>(x, x0f, N);
        gemm(x0f, bt_g[li], b_g[li], gbuf, N, 64, 64, 64, 1);              // g = silu(x0@w_g+b_g)
        k_gate<<<((N * 256) + 255) / 256, 256, 0, stream>>>(x, gbuf, h0, h13, N);
        gemm(h0,  bt_f0[li], nullptr, out0,  N,   64, 64, 64, 0);
        gemm(h13, bt_f1[li], nullptr, out13, N3v, 64, 64, 64, 0);
        k_add_out<<<((N * 256) + 255) / 256, 256, 0, stream>>>(x, out0, out13, N);
    };
    auto eupd = [&](int li) {
        int tot = E * 160;
        k_build_inv<<<(tot + 255) / 256, 256, 0, stream>>>(x, ef32, srcIdx, (_Float16*)bufA, N);
        gemm(bufA, bt_ew1[li], eb1[li], bufB, E, 64, 64, 160, 1 | 2);      // silu, f16
        gemm(bufB, bt_ew2[li], eb2[li], ef32, E, 32, 32, 64, 0);           // new ef f32
    };

    for (int li = 0; li < 4; ++li) { attn(li, 64); ffn(li); eupd(li); }

    attn(4, 32);   // mu
    k_final<<<((N * 128) + 255) / 256, 256, 0, stream>>>((float*)d_out, out0, out13, 0, N);
    attn(5, 32);   // logvar
    k_final<<<((N * 128) + 255) / 256, 256, 0, stream>>>((float*)d_out, out0, out13, 1, N);
}
